// SimpleRetention_10737418240742
// MI455X (gfx1250) — matrix operand order repaired and checked by reading, not compiled, not run on hardware
//
#include <hip/hip_runtime.h>
#include <hip/hip_bf16.h>
#include <cstdint>
#include <cstddef>

// ---------------------------------------------------------------------------
// SimpleRetention on MI455X (gfx1250): bf16 WMMA with fp32 accumulation.
//   scores = (Q xpos) @ (K xpos,downscale)^T * D,  out = scores @ V
// Round 2: native v_cvt bf16 conversions; async global->LDS staging for K.
// ---------------------------------------------------------------------------

typedef __attribute__((ext_vector_type(16))) __bf16 v16bf;
typedef __attribute__((ext_vector_type(8)))  float  v8f;

#define HIDDEN  256
#define HEAD    256
#define SEQ     2048
#define NHEADS  16                 // B*N = 2*8
#define MROWS   (NHEADS * SEQ)     // 32768 rows of X/Q/K/V

// Native conversion: lowers to v_cvt_{pk_}bf16_f32 on gfx1250 (RNE in HW).
__device__ __forceinline__ unsigned short f2bf(float f) {
  __bf16 h = (__bf16)f;
  return __builtin_bit_cast(unsigned short, h);
}

union V16 {                                   // one WMMA 16-bit operand (8 VGPRs)
  v16bf v;
  uint4 q[2];
  unsigned short s[16];
  __bf16 b[16];
};

__device__ __forceinline__ v8f wmma_bf16(v16bf a, v16bf b, v8f c) {
  return __builtin_amdgcn_wmma_f32_16x16x32_bf16(false, a, false, b,
                                                 (short)0, c, false, false);
}

// Wave-relative LDS byte offset of a generic pointer to __shared__ memory
// (generic LDS addresses carry the offset in the low 32 bits).
__device__ __forceinline__ unsigned lds_off(const void* p) {
  return (unsigned)(unsigned long long)p;
}

// Async copy of 16B/lane from global to LDS (tracked by ASYNCcnt).
__device__ __forceinline__ void async_copy_b128(void* lds_dst, const void* gsrc) {
  asm volatile("global_load_async_to_lds_b128 %0, %1, off"
               :: "v"(lds_off(lds_dst)),
                  "v"((unsigned long long)(size_t)gsrc)
               : "memory");
}

__device__ __forceinline__ void wait_asynccnt0() {
  asm volatile("s_wait_asynccnt 0x0" ::: "memory");
}

// ---------------------------------------------------------------------------
// Kernel 0: W (256x256 fp32, row-major [k][n]) -> WT (bf16, [n][k])
// ---------------------------------------------------------------------------
__global__ void k_wtrans(const float* __restrict__ Wq,
                         const float* __restrict__ Wk,
                         const float* __restrict__ Wv,
                         unsigned short* __restrict__ WT) {
  const int mat = blockIdx.y;
  const float* W = (mat == 0) ? Wq : (mat == 1) ? Wk : Wv;
  const int idx = blockIdx.x * blockDim.x + threadIdx.x;   // 0..65535
  const int n = idx >> 8;
  const int k = idx & 255;
  WT[mat * (HIDDEN * HEAD) + idx] = f2bf(W[k * HEAD + n]);
}

// ---------------------------------------------------------------------------
// Kernel 1: Q/K/V projection GEMM (bf16 WMMA, fp32 acc) + xpos rotary epilogue
// One wave computes one 16x16 output tile; K=256 contracted in 8 WMMAs.
// ---------------------------------------------------------------------------
__global__ void __launch_bounds__(256)
k_proj(const float* __restrict__ X,
       const unsigned short* __restrict__ WT,
       unsigned short* __restrict__ QKV) {
  const int lane = threadIdx.x & 31;
  const int wave = threadIdx.x >> 5;
  const bool hi  = lane >= 16;
  const int tile  = blockIdx.x * 8 + wave;   // 0..32767
  const int mtile = tile >> 4;               // 0..2047 (row tiles)
  const int ntile = tile & 15;               // 0..15   (col tiles)
  const int mat   = blockIdx.y;              // 0=Q, 1=K, 2=V

  const unsigned short* Wt = WT + (size_t)mat * (HIDDEN * HEAD);
  unsigned short* Out = QKV + (size_t)mat * MROWS * HEAD;

  const int arow = mtile * 16 + (lane & 15);

  v8f acc = {0.f, 0.f, 0.f, 0.f, 0.f, 0.f, 0.f, 0.f};
  for (int c = 0; c < 8; ++c) {
    // A operand (rows of X, fp32 -> bf16), documented 16-bit A 16x32 layout:
    //   lanes 0-15:  K = {c*32+0..7,  c*32+16..23}
    //   lanes 16-31: K = {c*32+8..15, c*32+24..31}
    V16 a;
    const float* xr = X + (size_t)arow * HIDDEN;
    const int o1 = c * 32 + (hi ? 8 : 0);
    const int o2 = o1 + 16;
    for (int j = 0; j < 8; ++j) {
      a.b[j]     = (__bf16)xr[o1 + j];
      a.b[8 + j] = (__bf16)xr[o2 + j];
    }
    // B operand from WT[n][k]: lane n holds k = {8h..8h+7, 16+8h..16+8h+7}
    V16 b;
    const unsigned short* wr =
        Wt + (size_t)(ntile * 16 + (lane & 15)) * HIDDEN + c * 32 + (hi ? 8 : 0);
    b.q[0] = *(const uint4*)(wr);
    b.q[1] = *(const uint4*)(wr + 16);
    acc = wmma_bf16(a.v, b.v, acc);
  }

  // Epilogue: xpos rotary for Q (mat 0) and K (mat 1, downscale), then bf16.
  const int col = ntile * 16 + (lane & 15);
  for (int r = 0; r < 8; ++r) {
    const int m = mtile * 16 + r + (hi ? 8 : 0);
    const int s = m & (SEQ - 1);               // position in sequence
    const float v = acc[r];
    float outv;
    if (mat < 2) {
      // adjacent channel partner lives in the adjacent lane (same M, N^1)
      const float partner = __shfl_xor(v, 1, 32);
      const int h = col >> 1;                  // half-index 0..127
      const float base_scale =
          ((float)(col & ~1) + 0.4f * (float)HEAD) / (1.4f * (float)HEAD);
      float e = (float)s * (1.0f / 512.0f);
      if (mat == 1) e = -e;                    // downscale for K
      const float scale    = __powf(base_scale, e);
      const float inv_freq = __powf(10000.0f, -(float)h * (1.0f / 128.0f));
      const float theta    = (float)s * inv_freq;
      const float sv = __sinf(theta) * scale;
      const float cv = __cosf(theta) * scale;
      outv = (col & 1) ? (v * cv + partner * sv)   // odd:  x2*cos + x1*sin
                       : (v * cv - partner * sv);  // even: x1*cos - x2*sin
    } else {
      outv = v;
    }
    Out[(size_t)m * HEAD + col] = f2bf(outv);
  }
}

// ---------------------------------------------------------------------------
// Kernel 2: causal decayed attention, streaming over 32-key blocks.
// Block = 256 threads = 8 waves; wave w owns 16 query rows. K block staged
// row-major in LDS with ASYNC global->LDS copies; V block staged transposed
// through VGPRs; score tiles re-quantized to bf16 through per-wave LDS
// scratch (C-layout -> A-layout).
// ---------------------------------------------------------------------------
__global__ void __launch_bounds__(256)
k_attn(const unsigned short* __restrict__ QKV, float* __restrict__ Out) {
  __shared__ __align__(16) unsigned short Kl[32 * 256];   // 16 KB
  __shared__ __align__(16) unsigned short Vt[256 * 32];   // 16 KB (transposed)
  __shared__ __align__(16) unsigned short Ss[8 * 16 * 32];// 8 KB score scratch

  const int lane = threadIdx.x & 31;
  const int wave = threadIdx.x >> 5;
  const bool hi  = lane >= 16;
  const int head = blockIdx.y;                 // 0..15
  const int qb   = blockIdx.x * 128;           // query block base
  const int q0   = qb + wave * 16;             // this wave's first query row

  const unsigned short* Qb = QKV;
  const unsigned short* Kb = QKV + (size_t)MROWS * HEAD;
  const unsigned short* Vb = QKV + (size_t)2 * MROWS * HEAD;
  const size_t hbase = (size_t)head * SEQ * HEAD;

  // Preload Q A-operands: 16 rows x 256 dims, 8 chunks of 16x32.
  v16bf qa[8];
  for (int c = 0; c < 8; ++c) {
    V16 t;
    const unsigned short* qr =
        Qb + hbase + (size_t)(q0 + (lane & 15)) * HEAD + c * 32 + (hi ? 8 : 0);
    t.q[0] = *(const uint4*)qr;
    t.q[1] = *(const uint4*)(qr + 16);
    qa[c] = t.v;
  }

  v8f o[16];
  const v8f vzero = {0.f, 0.f, 0.f, 0.f, 0.f, 0.f, 0.f, 0.f};
  for (int d = 0; d < 16; ++d) o[d] = vzero;

  const int nt = qb / 32 + 4;                  // causal: keys <= qb+127
  const float LNG = -0.0317486983f;            // ln(0.96875)

  for (int tb = 0; tb < nt; ++tb) {
    __syncthreads();                           // prior reads done before refill
    // Cooperative stage: K via async global->LDS (ASYNCcnt), V transposed
    // through VGPRs. 1024 16B chunks each, 256 threads.
    for (int i = 0; i < 4; ++i) {
      const int idx = threadIdx.x + i * 256;   // 0..1023
      const int r  = idx >> 5;                 // key row 0..31
      const int ch = idx & 31;                 // 8-element group
      const size_t grow = hbase + (size_t)(tb * 32 + r) * HEAD + ch * 8;
      async_copy_b128(Kl + r * 256 + ch * 8, Kb + grow);
      V16 t;
      t.q[0] = *(const uint4*)(Vb + grow);
      for (int j = 0; j < 8; ++j) Vt[(ch * 8 + j) * 32 + r] = t.s[j];
    }
    wait_asynccnt0();                          // drain async LDS writes
    __syncthreads();

    const bool active = (tb * 32) <= (q0 + 15);  // tile touches causal region
    if (active) {
      // ---- scores: 16 queries x 32 keys, contraction over 256 head dims ----
      v8f sc0 = vzero, sc1 = vzero;
      for (int c = 0; c < 8; ++c) {
        V16 b0, b1;
        const unsigned short* k0 =
            Kl + (size_t)(lane & 15) * 256 + c * 32 + (hi ? 8 : 0);
        const unsigned short* k1 =
            Kl + (size_t)(16 + (lane & 15)) * 256 + c * 32 + (hi ? 8 : 0);
        b0.q[0] = *(const uint4*)k0;  b0.q[1] = *(const uint4*)(k0 + 16);
        b1.q[0] = *(const uint4*)k1;  b1.q[1] = *(const uint4*)(k1 + 16);
        sc0 = wmma_bf16(qa[c], b0.v, sc0);
        sc1 = wmma_bf16(qa[c], b1.v, sc1);
      }
      // ---- exact per-element decay D = gamma^(s-t), causal mask ----
      unsigned short* sw = Ss + wave * 512;     // 16x32 bf16 scratch
      for (int r = 0; r < 8; ++r) {
        const int M = r + (hi ? 8 : 0);
        const int srow = q0 + M;
        const int t0 = tb * 32 + (lane & 15);
        const int t1 = t0 + 16;
        const float w0 = (srow >= t0) ? __expf((float)(srow - t0) * LNG) : 0.f;
        const float w1 = (srow >= t1) ? __expf((float)(srow - t1) * LNG) : 0.f;
        sw[M * 32 + (lane & 15)]      = f2bf(sc0[r] * w0);
        sw[M * 32 + 16 + (lane & 15)] = f2bf(sc1[r] * w1);
      }
    }
    __syncthreads();                           // scores visible (and waves sync)
    if (active) {
      // A operand: 16x32 bf16 scores in documented A layout
      V16 sa;
      const unsigned short* sr =
          Ss + wave * 512 + (size_t)(lane & 15) * 32 + (hi ? 8 : 0);
      sa.q[0] = *(const uint4*)sr;
      sa.q[1] = *(const uint4*)(sr + 16);
      // ---- O += S @ V : 16 dim-chunks of 16 cols each ----
      for (int d = 0; d < 16; ++d) {
        V16 b;
        const unsigned short* vr =
            Vt + (size_t)(d * 16 + (lane & 15)) * 32 + (hi ? 8 : 0);
        b.q[0] = *(const uint4*)vr;
        b.q[1] = *(const uint4*)(vr + 16);
        o[d] = wmma_bf16(sa.v, b.v, o[d]);
      }
    }
  }

  // Write fp32 output, layout (head, s, dim)
  float* op = Out + hbase;
  for (int d = 0; d < 16; ++d) {
    const int col = d * 16 + (lane & 15);
    for (int r = 0; r < 8; ++r) {
      const int m = q0 + r + (hi ? 8 : 0);
      op[(size_t)m * HEAD + col] = o[d][r];
    }
  }
}

// ---------------------------------------------------------------------------
extern "C" void kernel_launch(void* const* d_in, const int* in_sizes, int n_in,
                              void* d_out, int out_size, void* d_ws, size_t ws_size,
                              hipStream_t stream) {
  const float* X  = (const float*)d_in[0];
  const float* Wq = (const float*)d_in[1];
  const float* Wk = (const float*)d_in[2];
  const float* Wv = (const float*)d_in[3];

  unsigned short* ws  = (unsigned short*)d_ws;
  unsigned short* QKV = ws;                                   // 3*32768*256 bf16
  unsigned short* WT  = ws + (size_t)3 * MROWS * HEAD;        // 3*256*256 bf16

  // 0) transpose/convert weights
  k_wtrans<<<dim3(256, 3), 256, 0, stream>>>(Wq, Wk, Wv, WT);
  // 1) Q/K/V projections + xpos rotary (bf16 out)
  k_proj<<<dim3(4096, 3), 256, 0, stream>>>(X, WT, QKV);
  // 2) decayed causal attention
  k_attn<<<dim3(16, 16), 256, 0, stream>>>(QKV, (float*)d_out);
}
